// LocalFusionModule_3461743641056
// MI455X (gfx1250) — matrix operand order repaired and checked by reading, not compiled, not run on hardware
//
#include <hip/hip_runtime.h>
#include <stdint.h>

typedef __attribute__((ext_vector_type(16))) _Float16 v16h;
typedef __attribute__((ext_vector_type(8)))  float    v8f;

#define BB    8
#define NTOTC 8
#define NREF  7
#define CCH   128
#define HWN   4096
#define NQ    1024
#define QTILE 16
#define EPSF  1e-12f

// ---------------------------------------------------------------------------
// 1) inverse L2 norm of every kept ref column: invn[(b*7+n)*4096 + k]
// ---------------------------------------------------------------------------
__global__ void k_refnorm(const float* __restrict__ refs,
                          const int* __restrict__ dindex,
                          float* __restrict__ invn) {
    int t = blockIdx.x * blockDim.x + threadIdx.x;      // over B*NREF*HW
    if (t >= BB * NREF * HWN) return;
    int k  = t & (HWN - 1);
    int bn = t >> 12;
    int n  = bn % NREF;
    int b  = bn / NREF;
    int idx  = dindex[0];
    int keep = n + (n >= idx);
    const float* p = refs + ((size_t)(b * NTOTC + keep) * CCH) * HWN + k;
    float ss = 0.f;
#pragma unroll 8
    for (int c = 0; c < CCH; ++c) {
        float v = p[(size_t)c * HWN];
        ss += v * v;
    }
    invn[t] = 1.0f / fmaxf(sqrtf(ss), EPSF);
}

// ---------------------------------------------------------------------------
// 2) selected feat columns: normalized f16 (WMMA B layout: [b][c][q]) and
//    raw f32 feat_select ([b][q][c]).  One 128-thread block per (b,q).
// ---------------------------------------------------------------------------
__global__ void k_featprep(const float* __restrict__ feat,
                           const int* __restrict__ fidx,
                           _Float16* __restrict__ wfs,
                           float* __restrict__ fsel) {
    int bq = blockIdx.x;
    int q  = bq & (NQ - 1);
    int b  = bq >> 10;
    int c  = threadIdx.x;
    int pix = fidx[b * NQ + q];
    float v = feat[((size_t)(b * CCH + c)) * HWN + pix];
    __shared__ float red[128];
    red[c] = v * v;
    __syncthreads();
    for (int s = 64; s > 0; s >>= 1) {
        if (c < s) red[c] += red[c + s];
        __syncthreads();
    }
    float inv = 1.0f / fmaxf(sqrtf(red[0]), EPSF);
    wfs[((size_t)(b * CCH + c)) * NQ + q]  = (_Float16)(v * inv);
    fsel[((size_t)(b * NQ + q)) * CCH + c] = v;
}

// ---------------------------------------------------------------------------
// 3) main kernel: D = refs_tile(16k x 128c, f16) x wfeat(128c x 16q, f16)
//    via 4 chained v_wmma_f32_16x16x32_f16, inv-norm applied in epilogue,
//    fused running argmax over k=4096. 8 waves split k; LDS final reduce.
// ---------------------------------------------------------------------------
__global__ __launch_bounds__(256)
void k_simargmax(const float* __restrict__ refs,
                 const int* __restrict__ dindex,
                 const _Float16* __restrict__ wfs,
                 const float* __restrict__ invn,
                 int* __restrict__ ridx_ws,
                 int* __restrict__ ridx_out) {
    const int b  = blockIdx.z;
    const int n  = blockIdx.y;
    const int q0 = blockIdx.x * QTILE;
    const int lane = threadIdx.x & 31;
    const int wid  = threadIdx.x >> 5;
    const int r  = lane & 15;
    const int hi = lane >> 4;
    const int idx  = dindex[0];
    const int keep = n + (n >= idx);

    const float* refsb = refs + ((size_t)(b * NTOTC + keep) * CCH) * HWN;
    const float* invp  = invn + ((size_t)(b * NREF + n)) * HWN;

    // B operand: per lane q = q0 + r, element s -> channel c = cc*32 + (s>>3)*16 + hi*8 + (s&7)
    v16h bm[4];
#pragma unroll
    for (int cc = 0; cc < 4; ++cc) {
#pragma unroll
        for (int s = 0; s < 16; ++s) {
            const int c = cc * 32 + (s >> 3) * 16 + hi * 8 + (s & 7);
            bm[cc][s] = wfs[((size_t)(b * CCH + c)) * NQ + q0 + r];
        }
    }

    float best  = -3.4e38f;
    int   bestk = 0;

    for (int k0 = wid * 16; k0 < HWN; k0 += 128) {
        const int kcol = k0 + r;          // A row m = r -> ref column k0+r
        v8f acc = {};
#pragma unroll
        for (int cc = 0; cc < 4; ++cc) {
            v16h a;
#pragma unroll
            for (int h = 0; h < 16; ++h) {
                const int v = h >> 1;
                const int c = cc * 32 + (v >> 2) * 16 + hi * 8 + (v & 3) * 2 + (h & 1);
                a[h] = (_Float16)refsb[(size_t)c * HWN + kcol];
            }
            acc = __builtin_amdgcn_wmma_f32_16x16x32_f16(
                false, a, false, bm[cc], (short)0, acc, false, false);
        }
        // D tile: lane -> q = q0 + r ; VGPR j -> k = k0 + hi*8 + j
#pragma unroll
        for (int j = 0; j < 8; ++j) {
            const int   k = k0 + hi * 8 + j;
            const float v = acc[j] * invp[k];
            if (v > best || (v == best && k < bestk)) { best = v; bestk = k; }
        }
    }

    // combine the two half-lanes that share the same q
    float ob = __shfl_xor(best, 16, 32);
    int   ok = __shfl_xor(bestk, 16, 32);
    if (ob > best || (ob == best && ok < bestk)) { best = ob; bestk = ok; }

    __shared__ float sB[8 * 16];
    __shared__ int   sK[8 * 16];
    if (hi == 0) { sB[wid * 16 + r] = best; sK[wid * 16 + r] = bestk; }
    __syncthreads();

    if (threadIdx.x < 16) {
        float bv = sB[threadIdx.x];
        int   bk = sK[threadIdx.x];
#pragma unroll
        for (int w = 1; w < 8; ++w) {
            float v = sB[w * 16 + threadIdx.x];
            int   k = sK[w * 16 + threadIdx.x];
            if (v > bv || (v == bv && k < bk)) { bv = v; bk = k; }
        }
        const size_t o = ((size_t)(b * NREF + n)) * NQ + (q0 + threadIdx.x);
        ridx_ws[o]  = bk;
        ridx_out[o] = bk;   // third tuple-output region of d_out
    }
}

// ---------------------------------------------------------------------------
// 4) copy feat -> out, and emit feat_indices into d_out tail (int bits)
// ---------------------------------------------------------------------------
__global__ void k_copy(const float* __restrict__ feat,
                       const int* __restrict__ fidx,
                       float* __restrict__ out) {
    int i = blockIdx.x * blockDim.x + threadIdx.x;
    if (i < BB * CCH * HWN) out[i] = feat[i];
    if (i < BB * NQ) ((int*)(out + (size_t)BB * CCH * HWN))[i] = fidx[i];
}

// ---------------------------------------------------------------------------
// 5) fuse + scatter: out[b,c,pix] = base*feat_sel + sum_n sim_n * refs[...,argmax]
// ---------------------------------------------------------------------------
__global__ void k_fuse(const float* __restrict__ refs,
                       const float* __restrict__ sim,
                       const int* __restrict__ fidx,
                       const int* __restrict__ dindex,
                       const float* __restrict__ fsel,
                       const int* __restrict__ ridx,
                       float* __restrict__ out) {
    int bq = blockIdx.x;
    int q  = bq & (NQ - 1);
    int b  = bq >> 10;
    int c  = threadIdx.x;
    int idx = dindex[0];
    float val = sim[b * NTOTC + idx] * fsel[((size_t)(b * NQ + q)) * CCH + c];
#pragma unroll
    for (int j = 0; j < NREF; ++j) {
        int keep = j + (j >= idx);
        float s  = sim[b * NTOTC + keep];
        int   k  = ridx[((size_t)(b * NREF + j)) * NQ + q];
        val += s * refs[((size_t)(b * NTOTC + keep) * CCH + c) * HWN + k];
    }
    int pix = fidx[b * NQ + q];
    out[((size_t)(b * CCH + c)) * HWN + pix] = val;
}

// ---------------------------------------------------------------------------
extern "C" void kernel_launch(void* const* d_in, const int* in_sizes, int n_in,
                              void* d_out, int out_size, void* d_ws, size_t ws_size,
                              hipStream_t stream) {
    const float* feat   = (const float*)d_in[0];
    const float* refs   = (const float*)d_in[1];
    const float* sim    = (const float*)d_in[2];
    const int*   fidx   = (const int*)d_in[3];
    const int*   dindex = (const int*)d_in[4];
    float* out = (float*)d_out;

    char* ws = (char*)d_ws;
    size_t off = 0;
    float* invn = (float*)(ws + off);
    off += (size_t)BB * NREF * HWN * sizeof(float);   off = (off + 255) & ~(size_t)255;
    _Float16* wfs = (_Float16*)(ws + off);
    off += (size_t)BB * CCH * NQ * sizeof(_Float16);  off = (off + 255) & ~(size_t)255;
    float* fsel = (float*)(ws + off);
    off += (size_t)BB * NQ * CCH * sizeof(float);     off = (off + 255) & ~(size_t)255;
    int* ridx = (int*)(ws + off);

    // tuple output layout: [out 4194304 f32][feat_indices 8192 i32][ref_idx 57344 i32]
    int* ridx_out = (int*)(out + (size_t)BB * CCH * HWN + BB * NQ);

    k_refnorm<<<(BB * NREF * HWN + 255) / 256, 256, 0, stream>>>(refs, dindex, invn);
    k_featprep<<<BB * NQ, 128, 0, stream>>>(feat, fidx, wfs, fsel);

    dim3 grid(NQ / QTILE, NREF, BB);
    k_simargmax<<<grid, 256, 0, stream>>>(refs, dindex, wfs, invn, ridx, ridx_out);

    k_copy<<<(BB * CCH * HWN + 255) / 256, 256, 0, stream>>>(feat, fidx, out);
    k_fuse<<<BB * NQ, 128, 0, stream>>>(refs, sim, fidx, dindex, fsel, ridx, out);
}
